// E1DeepPredictor_12670153523362
// MI455X (gfx1250) — matrix operand order repaired and checked by reading, not compiled, not run on hardware
//
#include <hip/hip_runtime.h>
#include <hip/hip_bf16.h>
#include <math.h>

#define LATENT   512
#define HIDDEN   1024
#define SLOTS    16
#define BATCH    256
#define HORIZON_C 64

typedef __attribute__((ext_vector_type(16))) __bf16 v16bf;
typedef __attribute__((ext_vector_type(8)))  __bf16 v8bf;
typedef __attribute__((ext_vector_type(8)))  float  v8f;

// ---------------------------------------------------------------------------
// helpers
// ---------------------------------------------------------------------------
__device__ __forceinline__ v8f wmma_bf16(v16bf a, v16bf b, v8f c) {
  // 8 args: (neg_a, A, neg_b, B, c_mod, C, reuse_a, reuse_b)
  return __builtin_amdgcn_wmma_f32_16x16x32_bf16(false, a, false, b, (short)0, c,
                                                 false, false);
}

// A-fragment for v_wmma_f32_16x16x32_bf16 (row-major A, K-contiguous):
//   lanes 0-15  (row m): elems 0..7 = K[kk+0..7],  8..15 = K[kk+16..23]
//   lanes 16-31 (row m): elems 0..7 = K[kk+8..15], 8..15 = K[kk+24..31]
// caller passes p = Arow + kk + (lane>=16 ? 8 : 0)
__device__ __forceinline__ v16bf load_a_frag(const __bf16* p) {
  v8bf lo = *(const v8bf*)(p);
  v8bf hi = *(const v8bf*)(p + 16);
  v16bf r;
#pragma unroll
  for (int i = 0; i < 8; ++i) { r[i] = lo[i]; r[8 + i] = hi[i]; }
  return r;
}

__device__ __forceinline__ void bf16_split(float v, __bf16* ph, __bf16* pl) {
  __bf16 h = (__bf16)v;          // RNE
  float rem = v - (float)h;      // exact
  *ph = h;
  *pl = (__bf16)rem;
}

__device__ __forceinline__ float sigmoidf(float x) { return 1.0f / (1.0f + expf(-x)); }

// ---------------------------------------------------------------------------
// WMMA dual-source GEMM: C[M,N] = A1[M,K1]@W1[N,K1]^T + A2[M,K2]@W2[N,K2]^T + b1 + b2
// split-bf16 fp32 emulation: acc += Ah*Wh + Ah*Wl + Al*Wh
// one wave per 32x32 output tile (MT=2 x NT=2): 12 wmma per 16 b128 loads per
// K-chunk; halves L2 weight re-reads vs a 16-row tile.
// ---------------------------------------------------------------------------
__device__ __forceinline__ void gemm_source(const __bf16* __restrict__ Ah,
                                            const __bf16* __restrict__ Al, int K,
                                            const __bf16* __restrict__ Wh,
                                            const __bf16* __restrict__ Wl,
                                            int m0, int n0, int l16, int kh, v8f* acc) {
  const __bf16* a0h = Ah + (size_t)m0 * K + kh * 8;
  const __bf16* a0l = Al + (size_t)m0 * K + kh * 8;
  const __bf16* a1h = Ah + (size_t)(m0 + 16) * K + kh * 8;
  const __bf16* a1l = Al + (size_t)(m0 + 16) * K + kh * 8;
  const __bf16* w0h = Wh + (size_t)(n0 + l16) * K + kh * 8;
  const __bf16* w0l = Wl + (size_t)(n0 + l16) * K + kh * 8;
  const __bf16* w1h = Wh + (size_t)(n0 + 16 + l16) * K + kh * 8;
  const __bf16* w1l = Wl + (size_t)(n0 + 16 + l16) * K + kh * 8;
  for (int kk = 0; kk < K; kk += 32) {
    v16bf ah0 = load_a_frag(a0h + kk);
    v16bf al0 = load_a_frag(a0l + kk);
    v16bf ah1 = load_a_frag(a1h + kk);
    v16bf al1 = load_a_frag(a1l + kk);
    v16bf b0h = load_a_frag(w0h + kk);
    v16bf b0l = load_a_frag(w0l + kk);
    v16bf b1h = load_a_frag(w1h + kk);
    v16bf b1l = load_a_frag(w1l + kk);
    // m-tile 0
    acc[0] = wmma_bf16(ah0, b0h, acc[0]);
    acc[0] = wmma_bf16(ah0, b0l, acc[0]);
    acc[0] = wmma_bf16(al0, b0h, acc[0]);
    acc[1] = wmma_bf16(ah0, b1h, acc[1]);
    acc[1] = wmma_bf16(ah0, b1l, acc[1]);
    acc[1] = wmma_bf16(al0, b1h, acc[1]);
    // m-tile 1
    acc[2] = wmma_bf16(ah1, b0h, acc[2]);
    acc[2] = wmma_bf16(ah1, b0l, acc[2]);
    acc[2] = wmma_bf16(al1, b0h, acc[2]);
    acc[3] = wmma_bf16(ah1, b1h, acc[3]);
    acc[3] = wmma_bf16(ah1, b1l, acc[3]);
    acc[3] = wmma_bf16(al1, b1h, acc[3]);
  }
}

__global__ void __launch_bounds__(32)
wmma_dual_gemm(const __bf16* __restrict__ A1h, const __bf16* __restrict__ A1l, int K1,
               const __bf16* __restrict__ A2h, const __bf16* __restrict__ A2l, int K2,
               const __bf16* __restrict__ W1h, const __bf16* __restrict__ W1l,
               const __bf16* __restrict__ W2h, const __bf16* __restrict__ W2l,
               const float* __restrict__ bias1, const float* __restrict__ bias2,
               float* __restrict__ Cout, int ldc,
               __bf16* __restrict__ Oh, __bf16* __restrict__ Ol, int ldo,
               int relu) {
  const int lane = threadIdx.x & 31;
  const int kh   = lane >> 4;       // K-half selector (0: lanes 0-15, 1: lanes 16-31)
  const int l16  = lane & 15;
  const int n0   = blockIdx.x * 32; // two adjacent 16-wide N tiles
  const int m0   = blockIdx.y * 32 + l16;  // first of two 16-row M tiles

  v8f acc[4];
  acc[0] = (v8f){}; acc[1] = (v8f){}; acc[2] = (v8f){}; acc[3] = (v8f){};

  gemm_source(A1h, A1l, K1, W1h, W1l, m0, n0, l16, kh, acc);
  if (A2h) gemm_source(A2h, A2l, K2, W2h, W2l, m0, n0, l16, kh, acc);

  // C/D f32 layout: lanes 0-15 -> N=l16, rows tile+0..7 ; lanes 16-31 -> rows +8
#pragma unroll
  for (int s = 0; s < 2; ++s) {
    const int nglob = n0 + s * 16 + l16;
    float bv = 0.0f;
    if (bias1) bv += bias1[nglob];
    if (bias2) bv += bias2[nglob];
#pragma unroll
    for (int mi = 0; mi < 2; ++mi) {
      const int mbase = blockIdx.y * 32 + mi * 16 + kh * 8;
      const v8f a = acc[mi * 2 + s];
#pragma unroll
      for (int r = 0; r < 8; ++r) {
        float v = a[r] + bv;
        if (relu) v = fmaxf(v, 0.0f);
        Cout[(size_t)(mbase + r) * ldc + nglob] = v;
        if (Oh) bf16_split(v, &Oh[(size_t)(mbase + r) * ldo + nglob],
                              &Ol[(size_t)(mbase + r) * ldo + nglob]);
      }
    }
  }
}

// ---------------------------------------------------------------------------
// LSTM element-wise (PyTorch gate order i,f,g,o); writes c (fp32) and h as bf16 hi/lo
// ---------------------------------------------------------------------------
__global__ void lstm_pointwise(const float* __restrict__ gates, float* __restrict__ c,
                               __bf16* __restrict__ hh, __bf16* __restrict__ hl) {
  int idx = blockIdx.x * blockDim.x + threadIdx.x;   // BATCH*HIDDEN threads
  int b = idx >> 10, j = idx & 1023;
  const float* g = gates + (size_t)b * (4 * HIDDEN);
  float gi = sigmoidf(g[j]);
  float gf = sigmoidf(g[HIDDEN + j]);
  float gg = tanhf(g[2 * HIDDEN + j]);
  float go = sigmoidf(g[3 * HIDDEN + j]);
  float cn = gf * c[idx] + gi * gg;
  c[idx] = cn;
  float h = go * tanhf(cn);
  bf16_split(h, &hh[idx], &hl[idx]);
}

// ---------------------------------------------------------------------------
// prologue helpers (0.26% of FLOPs -> exact fp32 naive kernels)
// ---------------------------------------------------------------------------
__global__ void naive_linear(const float* __restrict__ A, const float* __restrict__ W,
                             const float* __restrict__ bias, float* __restrict__ C,
                             int M, int N, int K, int transB, float scale, int relu,
                             int ldc) {
  int idx = blockIdx.x * blockDim.x + threadIdx.x;
  if (idx >= M * N) return;
  int m = idx / N, n = idx % N;
  const float* a = A + (size_t)m * K;
  float acc = 0.0f;
  if (transB) {
    const float* w = W + (size_t)n * K;
    for (int k = 0; k < K; ++k) acc += a[k] * w[k];
  } else {
    for (int k = 0; k < K; ++k) acc += a[k] * W[(size_t)k * N + n];
  }
  float v = acc * scale + (bias ? bias[n] : 0.0f);
  if (relu) v = fmaxf(v, 0.0f);
  C[(size_t)m * ldc + n] = v;
}

__global__ void softmax16(float* __restrict__ s) {
  int r = blockIdx.x * blockDim.x + threadIdx.x;
  if (r >= BATCH) return;
  float* p = s + (size_t)r * SLOTS;
  float mx = p[0];
  for (int i = 1; i < SLOTS; ++i) mx = fmaxf(mx, p[i]);
  float sum = 0.0f;
  for (int i = 0; i < SLOTS; ++i) { float e = expf(p[i] - mx); p[i] = e; sum += e; }
  float inv = 1.0f / sum;
  for (int i = 0; i < SLOTS; ++i) p[i] *= inv;
}

__global__ void copy_to_combined(const float* __restrict__ cs, float* __restrict__ comb) {
  int idx = blockIdx.x * blockDim.x + threadIdx.x;
  if (idx >= BATCH * LATENT) return;
  int m = idx >> 9, n = idx & 511;
  comb[(size_t)m * (2 * LATENT) + n] = cs[idx];
}

__global__ void split_array(const float* __restrict__ X, __bf16* __restrict__ Xh,
                            __bf16* __restrict__ Xl, int n) {
  int i = blockIdx.x * blockDim.x + threadIdx.x;
  if (i < n) bf16_split(X[i], &Xh[i], &Xl[i]);
}

// ---------------------------------------------------------------------------
// launch
// ---------------------------------------------------------------------------
extern "C" void kernel_launch(void* const* d_in, const int* in_sizes, int n_in,
                              void* d_out, int out_size, void* d_ws, size_t ws_size,
                              hipStream_t stream) {
  const float* cs     = (const float*)d_in[0];
  const float* mem    = (const float*)d_in[2];
  const float* q_W    = (const float*)d_in[3];  const float* q_b   = (const float*)d_in[4];
  const float* k_W    = (const float*)d_in[5];  const float* k_b   = (const float*)d_in[6];
  const float* v_W    = (const float*)d_in[7];  const float* v_b   = (const float*)d_in[8];
  const float* mo_W   = (const float*)d_in[9];  const float* mo_b  = (const float*)d_in[10];
  const float* pg_W1  = (const float*)d_in[11]; const float* pg_b1 = (const float*)d_in[12];
  const float* pg_W2  = (const float*)d_in[13]; const float* pg_b2 = (const float*)d_in[14];
  const float* w_ih0  = (const float*)d_in[15]; const float* w_hh0 = (const float*)d_in[16];
  const float* b_ih0  = (const float*)d_in[17]; const float* b_hh0 = (const float*)d_in[18];
  const float* w_ih1  = (const float*)d_in[19]; const float* w_hh1 = (const float*)d_in[20];
  const float* b_ih1  = (const float*)d_in[21]; const float* b_hh1 = (const float*)d_in[22];
  const float* op_W1  = (const float*)d_in[23]; const float* op_b1 = (const float*)d_in[24];
  const float* op_W2  = (const float*)d_in[25]; const float* op_b2 = (const float*)d_in[26];
  float* out = (float*)d_out;
  const size_t PRED = (size_t)BATCH * HORIZON_C * LATENT;   // prior goes after predictions

  // workspace bump allocator (256B aligned)
  char* ws = (char*)d_ws;
  size_t off = 0;
  auto alloc = [&](size_t bytes) -> void* {
    void* p = ws + off;
    off = (off + bytes + 255) & ~(size_t)255;
    return p;
  };
  auto abf = [&](size_t elems) -> __bf16* { return (__bf16*)alloc(elems * sizeof(__bf16)); };
  auto af  = [&](size_t elems) -> float*  { return (float*)alloc(elems * sizeof(float)); };

  // bf16 hi/lo weight copies (resident in L2: ~62 MB)
  const size_t SZ_IH0 = (size_t)4 * HIDDEN * LATENT;
  const size_t SZ_HH  = (size_t)4 * HIDDEN * HIDDEN;
  const size_t SZ_OP1 = (size_t)HIDDEN * HIDDEN;
  const size_t SZ_OP2 = (size_t)LATENT * HIDDEN;
  __bf16 *wih0h = abf(SZ_IH0), *wih0l = abf(SZ_IH0);
  __bf16 *whh0h = abf(SZ_HH),  *whh0l = abf(SZ_HH);
  __bf16 *wih1h = abf(SZ_HH),  *wih1l = abf(SZ_HH);
  __bf16 *whh1h = abf(SZ_HH),  *whh1l = abf(SZ_HH);
  __bf16 *opW1h = abf(SZ_OP1), *opW1l = abf(SZ_OP1);
  __bf16 *opW2h = abf(SZ_OP2), *opW2l = abf(SZ_OP2);

  // activations
  __bf16 *xh = abf((size_t)BATCH * LATENT),  *xl = abf((size_t)BATCH * LATENT);
  __bf16 *h0h = abf((size_t)BATCH * HIDDEN), *h0l = abf((size_t)BATCH * HIDDEN);
  __bf16 *h1h = abf((size_t)BATCH * HIDDEN), *h1l = abf((size_t)BATCH * HIDDEN);
  __bf16 *hidh = abf((size_t)BATCH * HIDDEN), *hidl = abf((size_t)BATCH * HIDDEN);
  float *c0 = af((size_t)BATCH * HIDDEN), *c1 = af((size_t)BATCH * HIDDEN);
  float *gates = af((size_t)BATCH * 4 * HIDDEN);
  float *hidf  = af((size_t)BATCH * HIDDEN);
  // prologue buffers
  float *qbuf = af((size_t)BATCH * HIDDEN);
  float *kbuf = af((size_t)SLOTS * HIDDEN);
  float *vbuf = af((size_t)SLOTS * HIDDEN);
  float *scores = af((size_t)BATCH * SLOTS);
  float *ctxb = af((size_t)BATCH * HIDDEN);
  float *comb = af((size_t)BATCH * 2 * LATENT);
  float *pgh  = af((size_t)BATCH * HIDDEN);
  (void)ws_size; (void)in_sizes; (void)n_in; (void)out_size;

  // zero LSTM state (h hi/lo and c) every call — deterministic
  hipMemsetAsync(c0, 0, (size_t)BATCH * HIDDEN * sizeof(float), stream);
  hipMemsetAsync(c1, 0, (size_t)BATCH * HIDDEN * sizeof(float), stream);
  hipMemsetAsync(h0h, 0, (size_t)BATCH * HIDDEN * sizeof(__bf16), stream);
  hipMemsetAsync(h0l, 0, (size_t)BATCH * HIDDEN * sizeof(__bf16), stream);
  hipMemsetAsync(h1h, 0, (size_t)BATCH * HIDDEN * sizeof(__bf16), stream);
  hipMemsetAsync(h1l, 0, (size_t)BATCH * HIDDEN * sizeof(__bf16), stream);

  auto split = [&](const float* src, __bf16* dh, __bf16* dl, size_t n) {
    split_array<<<dim3((unsigned)((n + 255) / 256)), dim3(256), 0, stream>>>(src, dh, dl, (int)n);
  };
  split(w_ih0, wih0h, wih0l, SZ_IH0);
  split(w_hh0, whh0h, whh0l, SZ_HH);
  split(w_ih1, wih1h, wih1l, SZ_HH);
  split(w_hh1, whh1h, whh1l, SZ_HH);
  split(op_W1, opW1h, opW1l, SZ_OP1);
  split(op_W2, opW2h, opW2l, SZ_OP2);

  auto L = [&](const float* A, const float* W, const float* b, float* C, int M, int N,
               int K, int transB, float scale, int relu, int ldc) {
    int total = M * N;
    naive_linear<<<dim3((total + 255) / 256), dim3(256), 0, stream>>>(
        A, W, b, C, M, N, K, transB, scale, relu, ldc);
  };

  // --- ContextMemory.read + prior_generator (exact fp32) ---
  L(cs,  q_W, q_b, qbuf, BATCH, HIDDEN, LATENT, 1, 1.0f, 0, HIDDEN);
  L(mem, k_W, k_b, kbuf, SLOTS, HIDDEN, HIDDEN, 1, 1.0f, 0, HIDDEN);
  L(mem, v_W, v_b, vbuf, SLOTS, HIDDEN, HIDDEN, 1, 1.0f, 0, HIDDEN);
  L(qbuf, kbuf, nullptr, scores, BATCH, SLOTS, HIDDEN, 1, 0.03125f /*1/sqrt(1024)*/, 0, SLOTS);
  softmax16<<<dim3(1), dim3(256), 0, stream>>>(scores);
  L(scores, vbuf, nullptr, ctxb, BATCH, HIDDEN, SLOTS, 0, 1.0f, 0, HIDDEN);
  copy_to_combined<<<dim3((BATCH * LATENT + 255) / 256), dim3(256), 0, stream>>>(cs, comb);
  L(ctxb, mo_W, mo_b, comb + LATENT, BATCH, LATENT, HIDDEN, 1, 1.0f, 0, 2 * LATENT);
  L(comb, pg_W1, pg_b1, pgh, BATCH, HIDDEN, 2 * LATENT, 1, 1.0f, 1, HIDDEN);
  L(pgh,  pg_W2, pg_b2, out + PRED, BATCH, LATENT, HIDDEN, 1, 1.0f, 0, LATENT);
  split(out + PRED, xh, xl, (size_t)BATCH * LATENT);   // prior -> x for step 0

  // --- autoregressive rollout: 6 kernels per step, all WMMA GEMMs ---
  const dim3 blk(32);
  const int MROW = 32;                       // rows per wave (MT=2)
  for (int t = 0; t < HORIZON_C; ++t) {
    // layer 0 gates: x@w_ih0^T + h0@w_hh0^T + b_ih0 + b_hh0  -> [256,4096]
    wmma_dual_gemm<<<dim3(4 * HIDDEN / 32, BATCH / MROW), blk, 0, stream>>>(
        xh, xl, LATENT, h0h, h0l, HIDDEN,
        wih0h, wih0l, whh0h, whh0l, b_ih0, b_hh0,
        gates, 4 * HIDDEN, nullptr, nullptr, 0, 0);
    lstm_pointwise<<<dim3(BATCH * HIDDEN / 256), dim3(256), 0, stream>>>(gates, c0, h0h, h0l);
    // layer 1 gates: h0@w_ih1^T + h1@w_hh1^T + b
    wmma_dual_gemm<<<dim3(4 * HIDDEN / 32, BATCH / MROW), blk, 0, stream>>>(
        h0h, h0l, HIDDEN, h1h, h1l, HIDDEN,
        wih1h, wih1l, whh1h, whh1l, b_ih1, b_hh1,
        gates, 4 * HIDDEN, nullptr, nullptr, 0, 0);
    lstm_pointwise<<<dim3(BATCH * HIDDEN / 256), dim3(256), 0, stream>>>(gates, c1, h1h, h1l);
    // output_proj layer 1: relu(h1@op_W1^T + b) with bf16 side-output
    wmma_dual_gemm<<<dim3(HIDDEN / 32, BATCH / MROW), blk, 0, stream>>>(
        h1h, h1l, HIDDEN, nullptr, nullptr, 0,
        opW1h, opW1l, nullptr, nullptr, op_b1, nullptr,
        hidf, HIDDEN, hidh, hidl, HIDDEN, 1);
    // output_proj layer 2: pred -> d_out[:, t, :] and x hi/lo for next step
    wmma_dual_gemm<<<dim3(LATENT / 32, BATCH / MROW), blk, 0, stream>>>(
        hidh, hidl, HIDDEN, nullptr, nullptr, 0,
        opW2h, opW2l, nullptr, nullptr, op_b2, nullptr,
        out + (size_t)t * LATENT, HORIZON_C * LATENT, xh, xl, LATENT, 0);
  }
}
